// Interaction_36653250904505
// MI455X (gfx1250) — matrix operand order repaired and checked by reading, not compiled, not run on hardware
//
#include <hip/hip_runtime.h>
#include <hip/hip_bf16.h>

typedef __attribute__((ext_vector_type(16))) _Float16 v16h;
typedef __attribute__((ext_vector_type(8)))  _Float16 v8h;
typedef __attribute__((ext_vector_type(8)))  float    v8f;
typedef __attribute__((ext_vector_type(4)))  float    v4f;

#define DIM    64
#define HEADS  4
#define HD     16
#define HH     256
#define WW     256
#define NPIX   (HH * WW)
#define CORE_W 16
#define CORE_H 8
#define HALO_W 18
#define HALO_H 10
#define NHALO  (HALO_W * HALO_H)   // 180
#define NPAD   192                 // 12 N-tiles of 16
#define NCORE  (CORE_W * CORE_H)   // 128

// ---------------- WMMA helpers (gfx1250 wave32, v_wmma_f32_16x16x32_f16) ----

__device__ __forceinline__ v8f wmma_f16(v16h a, v16h b, v8f c) {
  return __builtin_amdgcn_wmma_f32_16x16x32_f16(
      /*neg_a=*/false, a, /*neg_b=*/false, b,
      /*c_mod=*/(short)0, c, /*reuse_a=*/false, /*reuse_b=*/false);
}

// A fragment from row-major LDS (row stride ld halves, 16B aligned runs):
// lane<16: row M=lane, K=kB..kB+15; lane>=16: K=kB+16..kB+31.
__device__ __forceinline__ v16h frag_a(const _Float16* lds, int ld, int mBase,
                                       int kBase, int lane) {
  int m = mBase + (lane & 15);
  int khi = (lane >= 16) ? 16 : 0;
  const _Float16* p = lds + m * ld + kBase + khi;
  v8h lo = *(const v8h*)p;        // K = kBase+khi+0..7
  v8h hi = *(const v8h*)(p + 8);  // K = kBase+khi+8..15
  return __builtin_shufflevector(lo, hi, 0, 1, 2, 3, 4, 5, 6, 7,
                                 8, 9, 10, 11, 12, 13, 14, 15);
}

// B fragment from packed-fragment LDS layout: [frag][lane][16] contiguous.
__device__ __forceinline__ v16h frag_b_packed(const _Float16* fragBase, int lane) {
  return *(const v16h*)(fragBase + lane * 16);
}

// B fragment from a transposed tile tileT[pixel][64ch]: K runs along channels.
// lane<16: N=col nBase+lane, K=cBase+0..15 ; lane>=16: N=nBase+lane-16, K=cBase+16..31
__device__ __forceinline__ v16h frag_b_T(const _Float16* tileT, int nBase,
                                         int cBase, int lane) {
  int n = nBase + (lane & 15);
  int koff = cBase + ((lane >= 16) ? 16 : 0);
  return *(const v16h*)(tileT + n * 64 + koff);
}

// ---------------- small kernels ---------------------------------------------

__global__ void k_zero(float* gram, float* qss, float* kss) {
  int i = blockIdx.x * 256 + threadIdx.x;
  if (i < 8 * HEADS * HD * HD) gram[i] = 0.0f;
  if (i < 8 * DIM) { qss[i] = 0.0f; kss[i] = 0.0f; }
}

// repack w_proj [o][c][ky][kx] -> f16 [o][pos*64 + c]  (implicit-GEMM K order)
__global__ void k_prep(const float* __restrict__ w_proj, _Float16* __restrict__ wp_h) {
  int i = blockIdx.x * 256 + threadIdx.x;
  if (i < DIM * DIM * 9) {
    int o = i / 576, rem = i % 576, pos = rem / 64, c = rem % 64;
    wp_h[o * 576 + pos * 64 + c] = (_Float16)w_proj[o * 576 + c * 9 + pos];
  }
}

// normalize gram by L2 norms, apply temperature, row softmax -> attn
__global__ __launch_bounds__(256)
void k_attn(const float* __restrict__ gram, const float* __restrict__ qss,
            const float* __restrict__ kss, const float* __restrict__ temp,
            float* __restrict__ attn) {
  __shared__ float sl[256];
  __shared__ float rmax[16], rsum[16];
  int bh = blockIdx.x, b = bh >> 2, h = bh & 3;
  int t = threadIdx.x, c = t >> 4, d = t & 15;
  float qn = fmaxf(sqrtf(qss[b * DIM + h * 16 + c]), 1e-12f);
  float kn = fmaxf(sqrtf(kss[b * DIM + h * 16 + d]), 1e-12f);
  float lg = gram[bh * 256 + t] / (qn * kn) * temp[h];
  sl[t] = lg;
  __syncthreads();
  if (t < 16) {
    float m = -1e30f;
    for (int j = 0; j < 16; ++j) m = fmaxf(m, sl[t * 16 + j]);
    rmax[t] = m;
  }
  __syncthreads();
  float e = __expf(sl[t] - rmax[c]);
  sl[t] = e;
  __syncthreads();
  if (t < 16) {
    float s = 0.0f;
    for (int j = 0; j < 16; ++j) s += sl[t * 16 + j];
    rsum[t] = s;
  }
  __syncthreads();
  attn[bh * 256 + t] = e / rsum[c];
}

// ---------------- fused LN + QKV GEMM + dwconv + attention stats ------------
// LDS: xnB packed-frag [24 frag][32 lane][16]h | w[64*64]h | dw[576]f |
//      rawT[192px][64ch]h | q[64][128]f | k[64][128]f
#define QKV_OFF_XN  0
#define QKV_OFF_W   24576
#define QKV_OFF_DW  (24576 + 8192)
#define QKV_OFF_RAW (24576 + 8192 + 2304)
#define QKV_OFF_Q   (24576 + 8192 + 2304 + 24576)
#define QKV_OFF_K   (24576 + 8192 + 2304 + 24576 + 32768)
#define QKV_LDS     (24576 + 8192 + 2304 + 24576 + 32768 + 32768)  // 125184

__global__ __launch_bounds__(256)
void k_qkv(const float* __restrict__ x, const float* __restrict__ ln_w,
           const float* __restrict__ ln_b, const float* __restrict__ wq,
           const float* __restrict__ wk, const float* __restrict__ wv,
           const float* __restrict__ dwq, const float* __restrict__ dwk,
           const float* __restrict__ dwv, float* __restrict__ vd,
           float* __restrict__ gram, float* __restrict__ qss,
           float* __restrict__ kss) {
  extern __shared__ unsigned char smem[];
  _Float16* lds_xnB  = (_Float16*)(smem + QKV_OFF_XN);
  _Float16* lds_w    = (_Float16*)(smem + QKV_OFF_W);
  float*    lds_dw   = (float*)(smem + QKV_OFF_DW);
  _Float16* lds_rawT = (_Float16*)(smem + QKV_OFF_RAW);
  float*    lds_q    = (float*)(smem + QKV_OFF_Q);
  float*    lds_k    = (float*)(smem + QKV_OFF_K);

  const int tid = threadIdx.x, lane = tid & 31, wv8 = tid >> 5;
  const int tileX = blockIdx.x, tileY = blockIdx.y, b = blockIdx.z;
  const int gx0 = tileX * CORE_W - 1, gy0 = tileY * CORE_H - 1;
  const float* xb = x + (size_t)b * DIM * NPIX;

  // zero packed xn region (covers halo-outside-image and N-tile padding)
  {
    v16h z = {};
    v16h* zp = (v16h*)lds_xnB;
    for (int i = tid; i < (DIM * NPAD) / 16; i += 256) zp[i] = z;
  }
  __syncthreads();

  // LayerNorm over channels at each halo pixel -> packed B-fragment layout
  for (int p = tid; p < NHALO; p += 256) {
    int hx = p % HALO_W, hy = p / HALO_W;
    int gx = gx0 + hx, gy = gy0 + hy;
    if (gx >= 0 && gx < WW && gy >= 0 && gy < HH) {
      int pix = gy * WW + gx;
      float xr[DIM];
      float s = 0.0f, ss = 0.0f;
#pragma unroll
      for (int c = 0; c < DIM; ++c) {
        float v = xb[(size_t)c * NPIX + pix];
        xr[c] = v; s += v; ss += v * v;
      }
      float mu = s * (1.0f / DIM);
      float var = ss * (1.0f / DIM) - mu * mu;
      float rs = rsqrtf(var + 1e-6f);
      v16h xv[4];
#pragma unroll
      for (int c = 0; c < DIM; ++c)
        xv[c >> 4][c & 15] = (_Float16)((xr[c] - mu) * rs * ln_w[c] + ln_b[c]);
      int pl = p & 15, nt = p >> 4;
      // fragment id = ks*12 + ntile ; lane = (p&15) + 16*(c16half)
      *(v16h*)(lds_xnB + ((0 * 12 + nt) * 32 + pl) * 16)      = xv[0];
      *(v16h*)(lds_xnB + ((0 * 12 + nt) * 32 + pl + 16) * 16) = xv[1];
      *(v16h*)(lds_xnB + ((1 * 12 + nt) * 32 + pl) * 16)      = xv[2];
      *(v16h*)(lds_xnB + ((1 * 12 + nt) * 32 + pl + 16) * 16) = xv[3];
    }
  }

  const float* wsrc[3] = {wq, wk, wv};
  const float* dsrc[3] = {dwq, dwk, dwv};
  for (int pr = 0; pr < 3; ++pr) {
    __syncthreads();
    const float* Wp = wsrc[pr];
    for (int i = tid; i < DIM * DIM; i += 256) lds_w[i] = (_Float16)Wp[i];
    const float* Dp = dsrc[pr];
    for (int i = tid; i < DIM * 9; i += 256) lds_dw[i] = Dp[i];
    __syncthreads();

    // GEMM: raw[64 x 192] = W[64x64] * xn[64x192]; D stored transposed
    for (int t = wv8; t < 48; t += 8) {
      int m = t & 3, n = t >> 2;
      v8f acc = {};
#pragma unroll
      for (int ks = 0; ks < 2; ++ks) {
        v16h a  = frag_a(lds_w, DIM, m * 16, ks * 32, lane);
        v16h bf = frag_b_packed(lds_xnB + (ks * 12 + n) * 512, lane);
        acc = wmma_f16(a, bf, acc);
      }
      int col = n * 16 + (lane & 15);
      int rbase = m * 16 + ((lane >= 16) ? 8 : 0);
      v8h pk;
#pragma unroll
      for (int r = 0; r < 8; ++r) pk[r] = (_Float16)acc[r];
      *(v8h*)(lds_rawT + col * 64 + rbase) = pk;  // one b128 store
    }
    __syncthreads();

    // depthwise 3x3: thread owns 8-channel block, taps cached in VGPRs
    {
      int cb = tid & 7, c0 = cb * 8;
      float wreg[8][9];
#pragma unroll
      for (int j = 0; j < 8; ++j)
#pragma unroll
        for (int pos = 0; pos < 9; ++pos)
          wreg[j][pos] = lds_dw[(c0 + j) * 9 + pos];
      for (int np = (tid >> 3); np < NCORE; np += 32) {
        int y = np >> 4, xc = np & 15;
        float accv[8] = {0, 0, 0, 0, 0, 0, 0, 0};
#pragma unroll
        for (int pos = 0; pos < 9; ++pos) {
          int ky = pos / 3, kx = pos % 3;
          v8h rv = *(const v8h*)(lds_rawT + ((y + ky) * HALO_W + xc + kx) * 64 + c0);
#pragma unroll
          for (int j = 0; j < 8; ++j) accv[j] += wreg[j][pos] * (float)rv[j];
        }
        if (pr == 0) {
#pragma unroll
          for (int j = 0; j < 8; ++j) lds_q[(c0 + j) * NCORE + np] = accv[j];
        } else if (pr == 1) {
#pragma unroll
          for (int j = 0; j < 8; ++j) lds_k[(c0 + j) * NCORE + np] = accv[j];
        } else {
          int gx = tileX * CORE_W + xc, gy = tileY * CORE_H + y;
          float* vb = vd + (size_t)b * DIM * NPIX + gy * WW + gx;
#pragma unroll
          for (int j = 0; j < 8; ++j) vb[(size_t)(c0 + j) * NPIX] = accv[j];
        }
      }
    }
  }
  __syncthreads();

  // per-channel sum of squares for the spatial L2 norms (q and k), b128 reads
  if (tid < 128) {
    int c = tid & 63;
    const v4f* src = (const v4f*)(((tid < 64) ? lds_q : lds_k) + c * NCORE);
    float s = 0.0f;
    for (int n4 = 0; n4 < NCORE / 4; ++n4) {
      v4f v = src[n4];
      s += v.x * v.x + v.y * v.y + v.z * v.z + v.w * v.w;
    }
    atomicAdd(((tid < 64) ? qss : kss) + b * DIM + c, s);
  }
  // Gram numerators: gram[b,h,c,d] += sum_n q[h*16+c,n] * k[h*16+d,n]
  for (int g = tid; g < HEADS * HD * HD; g += 256) {
    int h = g >> 8, cc = (g >> 4) & 15, d = g & 15;
    const v4f* qr = (const v4f*)(lds_q + (h * 16 + cc) * NCORE);
    const v4f* kr = (const v4f*)(lds_k + (h * 16 + d) * NCORE);
    float s = 0.0f;
    for (int n4 = 0; n4 < NCORE / 4; ++n4) {
      v4f a = qr[n4], bb = kr[n4];
      s += a.x * bb.x + a.y * bb.y + a.z * bb.z + a.w * bb.w;
    }
    atomicAdd(gram + ((size_t)(b * HEADS + h) * HD + cc) * HD + d, s);
  }
}

// ---------------- fused attn@V + 3x3 proj conv + residual + compress --------
// LDS: attn[4][16][32]h | wcomp[16][64]h | vdT[192px][64]h | o2T[192px][64]h |
//      wp[64][576]h | xnT[128px][64]f | o3T[128px][64]h
#define FIN_OFF_ATTN 0
#define FIN_OFF_WC   4096
#define FIN_OFF_VD   (4096 + 2048)
#define FIN_OFF_O2   (4096 + 2048 + 24576)
#define FIN_OFF_WP   (4096 + 2048 + 24576 + 24576)
#define FIN_OFF_XN   (4096 + 2048 + 24576 + 24576 + 73728)
#define FIN_OFF_O3   (4096 + 2048 + 24576 + 24576 + 73728 + 32768)
#define FIN_LDS      (4096 + 2048 + 24576 + 24576 + 73728 + 32768 + 16384) // 178176

__global__ __launch_bounds__(256)
void k_final(const float* __restrict__ x, const float* __restrict__ ln_w,
             const float* __restrict__ ln_b, const float* __restrict__ w_comp,
             const float* __restrict__ attn, const float* __restrict__ vd,
             const _Float16* __restrict__ wp_h, float* __restrict__ out) {
  extern __shared__ unsigned char smem[];
  _Float16* lds_attn = (_Float16*)(smem + FIN_OFF_ATTN);
  _Float16* lds_wc   = (_Float16*)(smem + FIN_OFF_WC);
  _Float16* lds_vdT  = (_Float16*)(smem + FIN_OFF_VD);
  _Float16* lds_o2T  = (_Float16*)(smem + FIN_OFF_O2);
  _Float16* lds_wp   = (_Float16*)(smem + FIN_OFF_WP);
  float*    lds_xnT  = (float*)(smem + FIN_OFF_XN);
  _Float16* lds_o3T  = (_Float16*)(smem + FIN_OFF_O3);

  const int tid = threadIdx.x, lane = tid & 31, wv8 = tid >> 5;
  const int tileX = blockIdx.x, tileY = blockIdx.y, b = blockIdx.z;
  const int gx0 = tileX * CORE_W - 1, gy0 = tileY * CORE_H - 1;

  // phase 1: zero vdT tile; stage attn (K padded 16->32) and w_comp as f16
  {
    v16h z = {};
    v16h* zp = (v16h*)lds_vdT;
    for (int i = tid; i < (DIM * NPAD) / 16; i += 256) zp[i] = z;
  }
  for (int i = tid; i < HEADS * 16 * 32; i += 256) {
    int h = i >> 9, rem = i & 511, c = rem >> 5, d = rem & 31;
    lds_attn[i] = (d < 16)
        ? (_Float16)attn[((size_t)(b * HEADS + h) * 16 + c) * 16 + d]
        : (_Float16)0.0f;
  }
  for (int i = tid; i < HD * DIM; i += 256) lds_wc[i] = (_Float16)w_comp[i];
  __syncthreads();

  // phase 2: vd halo tile -> transposed f16 LDS; w_proj -> LDS; xn at core
  const float* vdb = vd + (size_t)b * DIM * NPIX;
  for (int p = tid; p < NHALO; p += 256) {
    int hx = p % HALO_W, hy = p / HALO_W;
    int gx = gx0 + hx, gy = gy0 + hy;
    if (gx >= 0 && gx < WW && gy >= 0 && gy < HH) {
      int pix = gy * WW + gx;
#pragma unroll
      for (int c0 = 0; c0 < DIM; c0 += 8) {
        v8h pk;
#pragma unroll
        for (int j = 0; j < 8; ++j)
          pk[j] = (_Float16)vdb[(size_t)(c0 + j) * NPIX + pix];
        *(v8h*)(lds_vdT + p * 64 + c0) = pk;
      }
    }
  }
  for (int i = tid; i < DIM * 576; i += 256) lds_wp[i] = wp_h[i];
  const float* xb = x + (size_t)b * DIM * NPIX;
  if (tid < NCORE) {
    int y = tid >> 4, xc = tid & 15;
    int pix = (tileY * CORE_H + y) * WW + tileX * CORE_W + xc;
    float xr[DIM];
    float s = 0.0f, ss = 0.0f;
#pragma unroll
    for (int c = 0; c < DIM; ++c) {
      float v = xb[(size_t)c * NPIX + pix];
      xr[c] = v; s += v; ss += v * v;
    }
    float mu = s * (1.0f / DIM);
    float var = ss * (1.0f / DIM) - mu * mu;
    float rs = rsqrtf(var + 1e-6f);
    v4f xv[16];
#pragma unroll
    for (int c = 0; c < DIM; ++c)
      xv[c >> 2][c & 3] = (xr[c] - mu) * rs * ln_w[c] + ln_b[c];
    v4f* dst = (v4f*)(lds_xnT + tid * 64);
#pragma unroll
    for (int q4 = 0; q4 < 16; ++q4) dst[q4] = xv[q4];
  }
  __syncthreads();

  // phase 3: out2 = attn @ vd per head over the halo tile (K=16 WMMA)
  for (int t = wv8; t < HEADS * 12; t += 8) {
    int h = t & 3, n = t >> 2;
    v16h a = frag_a(lds_attn + h * 512, 32, 0, 0, lane);
    v16h bf = {};
    if (lane < 16)  // K=16..31 rows are zero -> upper lane half contributes 0
      bf = *(const v16h*)(lds_vdT + (n * 16 + lane) * 64 + h * 16);
    v8f acc = {};
    acc = wmma_f16(a, bf, acc);
    int col = n * 16 + (lane & 15);
    int rbase = h * 16 + ((lane >= 16) ? 8 : 0);
    v8h pk;
#pragma unroll
    for (int r = 0; r < 8; ++r) pk[r] = (_Float16)acc[r];
    *(v8h*)(lds_o2T + col * 64 + rbase) = pk;
  }
  __syncthreads();

  // phase 4: 3x3 proj conv as implicit GEMM, K = 9 taps * 64 ch = 18 k-steps.
  // wave wv8 owns core row y = wv8 (one 16-wide N-tile), all 4 M-tiles.
  {
    int y = wv8, xc = lane & 15;
    v8f accs[4] = {{}, {}, {}, {}};
    for (int kk = 0; kk < 18; ++kk) {
      int pos = kk >> 1, ch = (kk & 1) * 32;
      int ky = pos / 3, kx = pos % 3;
      int hidx = (y + ky) * HALO_W + xc + kx;
      int koff = ch + ((lane >= 16) ? 16 : 0);
      v16h bf = *(const v16h*)(lds_o2T + hidx * 64 + koff);  // 2x b128
#pragma unroll
      for (int m = 0; m < 4; ++m) {
        v16h a = frag_a(lds_wp, 576, m * 16, kk * 32, lane);
        accs[m] = wmma_f16(a, bf, accs[m]);
      }
    }
    // add residual xn (b128 reads), stage out3 transposed f16
    int col = wv8 * 16 + (lane & 15);
#pragma unroll
    for (int m = 0; m < 4; ++m) {
      int rbase = m * 16 + ((lane >= 16) ? 8 : 0);
      v4f r0 = *(const v4f*)(lds_xnT + col * 64 + rbase);
      v4f r1 = *(const v4f*)(lds_xnT + col * 64 + rbase + 4);
      v8h pk;
#pragma unroll
      for (int r = 0; r < 4; ++r) pk[r] = (_Float16)(accs[m][r] + r0[r]);
#pragma unroll
      for (int r = 0; r < 4; ++r) pk[r + 4] = (_Float16)(accs[m][r + 4] + r1[r]);
      *(v8h*)(lds_o3T + col * 64 + rbase) = pk;
    }
  }
  __syncthreads();

  // phase 5: 1x1 compress 64->16 and write output
  {
    v8f acc = {};
#pragma unroll
    for (int ks = 0; ks < 2; ++ks) {
      v16h a  = frag_a(lds_wc, DIM, 0, ks * 32, lane);
      v16h bf = frag_b_T(lds_o3T, wv8 * 16, ks * 32, lane);
      acc = wmma_f16(a, bf, acc);
    }
    int xc = lane & 15;
    int gx = tileX * CORE_W + xc;
    int gy = tileY * CORE_H + wv8;
    int rbase = (lane >= 16) ? 8 : 0;
    float* ob = out + (size_t)b * HD * NPIX;
#pragma unroll
    for (int r = 0; r < 8; ++r)
      ob[(size_t)(rbase + r) * NPIX + gy * WW + gx] = acc[r];
  }
}

// ---------------- host launch -----------------------------------------------

extern "C" void kernel_launch(void* const* d_in, const int* in_sizes, int n_in,
                              void* d_out, int out_size, void* d_ws,
                              size_t ws_size, hipStream_t stream) {
  (void)in_sizes; (void)n_in; (void)out_size; (void)ws_size;
  const float* x      = (const float*)d_in[0];
  const float* ln_w   = (const float*)d_in[1];
  const float* ln_b   = (const float*)d_in[2];
  const float* w_q    = (const float*)d_in[3];
  const float* w_k    = (const float*)d_in[4];
  const float* w_v    = (const float*)d_in[5];
  const float* dw_q   = (const float*)d_in[6];
  const float* dw_k   = (const float*)d_in[7];
  const float* dw_v   = (const float*)d_in[8];
  const float* w_proj = (const float*)d_in[9];
  const float* w_comp = (const float*)d_in[10];
  const float* temp   = (const float*)d_in[11];
  float* out = (float*)d_out;

  // workspace carve (256B aligned blocks): vd | gram | qss | kss | attn | wp_h
  unsigned char* ws = (unsigned char*)d_ws;
  float*    vd    = (float*)ws;                                // 134217728 B
  float*    gram  = (float*)(ws + 134217728ull);               // 32768 B
  float*    qss   = (float*)(ws + 134217728ull + 32768);       // 2048 B
  float*    kss   = (float*)(ws + 134217728ull + 32768 + 2048);
  float*    attnb = (float*)(ws + 134217728ull + 32768 + 4096);
  _Float16* wp_h  = (_Float16*)(ws + 134217728ull + 32768 + 4096 + 32768);

  // allow >64KB dynamic LDS (gfx1250 WGP: up to 320KB)
  (void)hipFuncSetAttribute((const void*)k_qkv,
      hipFuncAttributeMaxDynamicSharedMemorySize, QKV_LDS);
  (void)hipFuncSetAttribute((const void*)k_final,
      hipFuncAttributeMaxDynamicSharedMemorySize, FIN_LDS);

  dim3 tiles(WW / CORE_W, HH / CORE_H, 8);  // (16, 32, 8)

  k_zero<<<32, 256, 0, stream>>>(gram, qss, kss);
  k_prep<<<(DIM * DIM * 9 + 255) / 256, 256, 0, stream>>>(w_proj, wp_h);
  k_qkv<<<tiles, 256, QKV_LDS, stream>>>(x, ln_w, ln_b, w_q, w_k, w_v, dw_q,
                                         dw_k, dw_v, vd, gram, qss, kss);
  k_attn<<<32, 256, 0, stream>>>(gram, qss, kss, temp, attnb);
  k_final<<<tiles, 256, FIN_LDS, stream>>>(x, ln_w, ln_b, w_comp, attnb, vd,
                                           wp_h, out);
}
